// MultiHeadAttention_5669356830737
// MI455X (gfx1250) — matrix operand order repaired and checked by reading, not compiled, not run on hardware
//
#include <hip/hip_runtime.h>

typedef __bf16 bf16_t;
typedef __bf16 v16bf __attribute__((ext_vector_type(16)));
typedef __bf16 v8bf  __attribute__((ext_vector_type(8)));
typedef float  v8f   __attribute__((ext_vector_type(8)));

#define DM 1024   // model dim
#define TT 2048   // seq len
#define NH 16     // heads
#define DH 64     // head dim
#define NB 2      // batch

static __device__ __forceinline__ v16bf cat8(v8bf lo, v8bf hi) {
  return __builtin_shufflevector(lo, hi, 0,1,2,3,4,5,6,7,8,9,10,11,12,13,14,15);
}
static __device__ __forceinline__ v8f wmma_bf16(v16bf a, v16bf b, v8f c) {
  // D = A(16x32) * B(32x16) + C, fp32 accumulate
  return __builtin_amdgcn_wmma_f32_16x16x32_bf16(false, a, false, b, (short)0, c, false, false);
}
static __device__ __forceinline__ v8f vzero8() { v8f z = {0.f,0.f,0.f,0.f,0.f,0.f,0.f,0.f}; return z; }

// Async copy 16B global -> LDS (per lane), tracked by ASYNCcnt.
// LDS address = low 32 bits of the flat shared pointer (LDS aperture keeps
// the offset in addr[31:0], ISA 10.2).
static __device__ __forceinline__ void async_copy16(bf16_t* lds_dst, const bf16_t* gsrc) {
  unsigned loff = (unsigned)(unsigned long long)lds_dst;
  asm volatile("global_load_async_to_lds_b128 %0, %1, off"
               :: "v"(loff), "v"(gsrc) : "memory");
}
static __device__ __forceinline__ void wait_asynccnt0() {
  asm volatile("s_wait_asynccnt 0x0" ::: "memory");
}

// ---------------- fp32 -> bf16 convert ----------------
__global__ void f32_to_bf16(const float* __restrict__ in, bf16_t* __restrict__ out, int n) {
  for (int i = blockIdx.x * blockDim.x + threadIdx.x; i < n; i += gridDim.x * blockDim.x)
    out[i] = (bf16_t)in[i];
}

// ---------------- fused QKV projection + RoPE ----------------
// out = x @ W^T for W in {wq,wk,wv}. Each wave: 16(M) x 64(N) tile, K=1024.
// q,k (rope'd) stored bf16 [B,H,T,dh]; v stored bf16 transposed [B,H,dh,T].
__global__ void __launch_bounds__(256) proj_qkv(
    const bf16_t* __restrict__ xb,
    const bf16_t* __restrict__ wqb, const bf16_t* __restrict__ wkb, const bf16_t* __restrict__ wvb,
    bf16_t* __restrict__ qo, bf16_t* __restrict__ ko, bf16_t* __restrict__ vto)
{
  const int lane = threadIdx.x & 31;
  const int wave = threadIdx.x >> 5;
  const int tile = blockIdx.x * 8 + wave;     // 256 M-tiles * 16 N-tiles = 4096
  const int m0 = (tile >> 4) * 16;
  const int n0 = (tile & 15) * 64;
  const int l16 = lane & 15, hi = lane >> 4;
  const int arow = m0 + l16;                  // A layout: lane = row (both halves)

  v8f acc[3][4];
#pragma unroll
  for (int x = 0; x < 3; ++x)
#pragma unroll
    for (int s = 0; s < 4; ++s) acc[x][s] = vzero8();

#pragma unroll 2
  for (int k = 0; k < DM; k += 32) {
    const bf16_t* ap = xb + (size_t)arow * DM + k + hi * 16;      // two 16B runs per lane
    v16bf A = cat8(*(const v8bf*)ap, *(const v8bf*)(ap + 8));
#pragma unroll
    for (int s = 0; s < 4; ++s) {
      const size_t woff = (size_t)(n0 + s * 16 + l16) * DM + k + hi * 16;  // B: lane = col
      acc[0][s] = wmma_bf16(A, *(const v16bf*)(wqb + woff), acc[0][s]);
      acc[1][s] = wmma_bf16(A, *(const v16bf*)(wkb + woff), acc[1][s]);
      acc[2][s] = wmma_bf16(A, *(const v16bf*)(wvb + woff), acc[2][s]);
    }
  }

  // RoPE on q,k (pairs are adjacent lanes in C layout), then scatter stores.
#pragma unroll
  for (int s = 0; s < 4; ++s) {
    const int n = n0 + s * 16 + l16;          // global out column
    const int h = n >> 6;
    const int dloc = n & 63;
    const bool even = (dloc & 1) == 0;
    const float freq = __powf(10000.f, -(float)(dloc >> 1) * (1.f / 32.f));
#pragma unroll
    for (int v = 0; v < 8; ++v) {
      const int m = m0 + v + hi * 8;          // global row (token)
      const int bidx = m >> 11;               // T = 2048
      const int trow = m & (TT - 1);
      float sn, cs;
      __sincosf((float)trow * freq, &sn, &cs);
      float qv = acc[0][s][v];
      float qp = __shfl_xor(qv, 1);
      float qr = even ? (qv * cs - qp * sn) : (qp * sn + qv * cs);
      float kv = acc[1][s][v];
      float kp = __shfl_xor(kv, 1);
      float kr = even ? (kv * cs - kp * sn) : (kp * sn + kv * cs);
      const size_t bh = (size_t)(bidx * NH + h);
      qo[(bh * TT + trow) * DH + dloc] = (bf16_t)qr;
      ko[(bh * TT + trow) * DH + dloc] = (bf16_t)kr;
      vto[(bh * DH + dloc) * TT + trow] = (bf16_t)acc[2][s][v];
    }
  }
}

// ---------------- flash attention (causal) ----------------
// One block (8 waves) handles one (b,h) and a 128-query span; wave w owns the
// 16-query tile q0 = qbase + 16w. K/V 32-key blocks are staged into LDS once
// per block via async global->LDS copies (ASYNCcnt) and shared by all waves.
__global__ void __launch_bounds__(256) attn(
    const bf16_t* __restrict__ qb, const bf16_t* __restrict__ kb,
    const bf16_t* __restrict__ vtb, bf16_t* __restrict__ ab)
{
  __shared__ __attribute__((aligned(128))) bf16_t klds[32 * 64];      // 4KB: [key][d]
  __shared__ __attribute__((aligned(128))) bf16_t vlds[64 * 32];      // 4KB: [d][key]
  __shared__ __attribute__((aligned(64)))  bf16_t plds[8 * 16 * 32];  // 8KB P bounce
  const int tid = threadIdx.x;
  const int lane = tid & 31;
  const int wave = tid >> 5;
  const int bh = blockIdx.x >> 4;             // B*H = 32
  const int qbase = (blockIdx.x & 15) * 128;  // 16 query chunks of 128
  const int q0 = qbase + wave * 16;
  const int l16 = lane & 15, hi = lane >> 4;

  const bf16_t* qp = qb  + (size_t)bh * TT * DH;
  const bf16_t* kp = kb  + (size_t)bh * TT * DH;
  const bf16_t* vp = vtb + (size_t)bh * DH * TT;

  // cooperative-load coordinates (256 threads move 4KB per matrix per block)
  const int krow = tid >> 3, kcol = (tid & 7) * 8;   // K: 32 rows x 64 cols
  const int vrow = tid >> 2, vcol = (tid & 3) * 8;   // V: 64 rows x 32 cols

  // Q as two A-tiles (K = d-chunks of 32)
  v16bf QA[2];
#pragma unroll
  for (int c = 0; c < 2; ++c) {
    const bf16_t* p = qp + (size_t)(q0 + l16) * DH + 32 * c + hi * 16;
    QA[c] = cat8(*(const v8bf*)p, *(const v8bf*)(p + 8));
  }

  v8f O[4]; float mrow[8], lrow[8];
#pragma unroll
  for (int s = 0; s < 4; ++s) O[s] = vzero8();
#pragma unroll
  for (int v = 0; v < 8; ++v) { mrow[v] = -3.0e38f; lrow[v] = 0.f; }

  const int jend = qbase + 128;               // block-wide causal bound
  for (int j = 0; j < jend; j += 32) {
    // ---- stage K/V block into LDS (async, no VGPR round trip) ----
    async_copy16(klds + krow * 64 + kcol, kp + (size_t)(j + krow) * DH + kcol);
    async_copy16(vlds + vrow * 32 + vcol, vp + (size_t)vrow * TT + j + vcol);
    wait_asynccnt0();
    __syncthreads();

    if (j <= q0) {  // this wave still has unmasked keys in [j, j+31]
      // S = Q K^T for 32 keys (two 16-key groups), K tiles from LDS
      v8f S0 = vzero8(), S1 = vzero8();
#pragma unroll
      for (int c = 0; c < 2; ++c) {
        S0 = wmma_bf16(QA[c], *(const v16bf*)(klds + (l16) * 64 + 32 * c + hi * 16), S0);
        S1 = wmma_bf16(QA[c], *(const v16bf*)(klds + (16 + l16) * 64 + 32 * c + hi * 16), S1);
      }
      // scale + causal mask
      const int key0 = j + l16, key1 = key0 + 16;
#pragma unroll
      for (int v = 0; v < 8; ++v) {
        const int qrow = q0 + v + hi * 8;
        float s0 = S0[v] * 0.125f; if (key0 > qrow) s0 = -3.0e38f;
        float s1 = S1[v] * 0.125f; if (key1 > qrow) s1 = -3.0e38f;
        S0[v] = s0; S1[v] = s1;
      }
      // online softmax: row reductions across each 16-lane half
      float alpha[8];
#pragma unroll
      for (int v = 0; v < 8; ++v) {
        float mx = fmaxf(S0[v], S1[v]);
        mx = fmaxf(mx, __shfl_xor(mx, 1));
        mx = fmaxf(mx, __shfl_xor(mx, 2));
        mx = fmaxf(mx, __shfl_xor(mx, 4));
        mx = fmaxf(mx, __shfl_xor(mx, 8));
        const float mn = fmaxf(mrow[v], mx);
        alpha[v] = __expf(mrow[v] - mn);
        mrow[v] = mn;
        const float p0 = __expf(S0[v] - mn), p1 = __expf(S1[v] - mn);
        S0[v] = p0; S1[v] = p1;
        float r = p0 + p1;
        r += __shfl_xor(r, 1);
        r += __shfl_xor(r, 2);
        r += __shfl_xor(r, 4);
        r += __shfl_xor(r, 8);
        lrow[v] = lrow[v] * alpha[v] + r;
      }
#pragma unroll
      for (int s = 0; s < 4; ++s)
#pragma unroll
        for (int v = 0; v < 8; ++v) O[s][v] *= alpha[v];

      // C-layout P -> A-layout bf16 via per-wave LDS bounce
      bf16_t* pl = plds + wave * 512;
#pragma unroll
      for (int v = 0; v < 8; ++v) {
        const int row = v + hi * 8;
        pl[row * 32 + l16]      = (bf16_t)S0[v];
        pl[row * 32 + 16 + l16] = (bf16_t)S1[v];
      }
      __builtin_amdgcn_wave_barrier();
      const bf16_t* pr = pl + l16 * 32 + hi * 16;
      v16bf PA = cat8(*(const v8bf*)pr, *(const v8bf*)(pr + 8));

      // O += P @ V, V tiles from LDS ([d][key] => B columns contiguous)
#pragma unroll
      for (int s = 0; s < 4; ++s) {
        const v16bf VB = *(const v16bf*)(vlds + (16 * s + l16) * 32 + hi * 16);
        O[s] = wmma_bf16(PA, VB, O[s]);
      }
    }
    __syncthreads();  // protect LDS tiles from next iteration's overwrite
  }

  // normalize + store merged-head bf16 [B,T,D]
  const int b = bh >> 4, h = bh & 15;
#pragma unroll
  for (int s = 0; s < 4; ++s)
#pragma unroll
    for (int v = 0; v < 8; ++v) {
      const int trow = q0 + v + hi * 8;
      const float o = O[s][v] / lrow[v];
      ab[((size_t)b * TT + trow) * DM + h * DH + 16 * s + l16] = (bf16_t)o;
    }
}

// ---------------- output projection ----------------
__global__ void __launch_bounds__(256) out_proj(
    const bf16_t* __restrict__ ab, const bf16_t* __restrict__ wob, float* __restrict__ out)
{
  const int lane = threadIdx.x & 31;
  const int wave = threadIdx.x >> 5;
  const int tile = blockIdx.x * 8 + wave;
  const int m0 = (tile >> 4) * 16;
  const int n0 = (tile & 15) * 64;
  const int l16 = lane & 15, hi = lane >> 4;
  const int arow = m0 + l16;

  v8f acc[4];
#pragma unroll
  for (int s = 0; s < 4; ++s) acc[s] = vzero8();

#pragma unroll 2
  for (int k = 0; k < DM; k += 32) {
    const bf16_t* ap = ab + (size_t)arow * DM + k + hi * 16;
    v16bf A = cat8(*(const v8bf*)ap, *(const v8bf*)(ap + 8));
#pragma unroll
    for (int s = 0; s < 4; ++s) {
      const size_t woff = (size_t)(n0 + s * 16 + l16) * DM + k + hi * 16;
      acc[s] = wmma_bf16(A, *(const v16bf*)(wob + woff), acc[s]);
    }
  }
#pragma unroll
  for (int s = 0; s < 4; ++s)
#pragma unroll
    for (int v = 0; v < 8; ++v)
      out[(size_t)(m0 + v + hi * 8) * DM + n0 + s * 16 + l16] = acc[s][v];
}

extern "C" void kernel_launch(void* const* d_in, const int* in_sizes, int n_in,
                              void* d_out, int out_size, void* d_ws, size_t ws_size,
                              hipStream_t stream) {
  const float* x  = (const float*)d_in[0];
  const float* wq = (const float*)d_in[1];
  const float* wk = (const float*)d_in[2];
  const float* wv = (const float*)d_in[3];
  const float* wo = (const float*)d_in[4];

  const size_t MB = 1u << 20;
  char* ws = (char*)d_ws;
  bf16_t* xb  = (bf16_t*)(ws + 0);          // 8 MB  (2*2048*1024)
  bf16_t* wqb = (bf16_t*)(ws + 8  * MB);    // 2 MB
  bf16_t* wkb = (bf16_t*)(ws + 10 * MB);
  bf16_t* wvb = (bf16_t*)(ws + 12 * MB);
  bf16_t* wob = (bf16_t*)(ws + 14 * MB);
  bf16_t* qb  = (bf16_t*)(ws + 16 * MB);    // 8 MB [B,H,T,dh]
  bf16_t* kbf = (bf16_t*)(ws + 24 * MB);    // 8 MB [B,H,T,dh]
  bf16_t* vtb = (bf16_t*)(ws + 32 * MB);    // 8 MB [B,H,dh,T]
  bf16_t* abf = (bf16_t*)(ws + 40 * MB);    // 8 MB [B,T,D]

  const int nx = NB * TT * DM;  // 4,194,304
  const int nw = DM * DM;       // 1,048,576
  f32_to_bf16<<<2048, 256, 0, stream>>>(x,  xb,  nx);
  f32_to_bf16<<<1024, 256, 0, stream>>>(wq, wqb, nw);
  f32_to_bf16<<<1024, 256, 0, stream>>>(wk, wkb, nw);
  f32_to_bf16<<<1024, 256, 0, stream>>>(wv, wvb, nw);
  f32_to_bf16<<<1024, 256, 0, stream>>>(wo, wob, nw);

  proj_qkv<<<512, 256, 0, stream>>>(xb, wqb, wkb, wvb, qb, kbf, vtb);
  attn<<<512, 256, 0, stream>>>(qb, kbf, vtb, abf);
  out_proj<<<512, 256, 0, stream>>>(abf, wob, (float*)d_out);
}
